// VitDetLayer_46308337385975
// MI455X (gfx1250) — compile-verified, matrix operand order checked by reading, not run on hardware
//
#include <hip/hip_runtime.h>
#include <hip/hip_bf16.h>
#include <math.h>
#include <stdint.h>

// ---------------- problem constants ----------------
#define DIMC   768
#define HEADS  12
#define HD     64
#define WSZ    14
#define NTOK   196      // tokens per window (14*14)
#define NPAD   208      // padded query rows (13*16)
#define KPAD   224      // padded key dim for P@V (7*32)
#define NWIN   100      // 4 * 5 * 5 windows
#define NBH    1200     // NWIN * HEADS
#define MQKV   19600    // NWIN * NTOK
#define MMLP   16384    // 4*64*64 valid tokens
#define MLPD   3072
#define EPSLN  1e-6f
#define SCALE  0.125f   // 64^-0.5

typedef __bf16 bf16;
typedef __attribute__((ext_vector_type(8)))  bf16  bf16x8;
typedef __attribute__((ext_vector_type(16))) bf16  bf16x16;
typedef __attribute__((ext_vector_type(8)))  float f32x8;
typedef __attribute__((ext_vector_type(4)))  unsigned int u32x4;
typedef __attribute__((ext_vector_type(8)))  int   i32x8;
typedef __attribute__((ext_vector_type(4)))  int   i32x4;

// ---- Tensor Data Mover availability (device pass only; host pass falls back) ----
#if defined(__has_builtin)
#if __has_builtin(__builtin_amdgcn_tensor_load_to_lds) && \
    __has_builtin(__builtin_amdgcn_s_wait_tensorcnt)
#define HAVE_TDM 1
#endif
#endif
#ifndef HAVE_TDM
#define HAVE_TDM 0
#endif

// ---------------- WMMA helpers ----------------
__device__ __forceinline__ f32x8 wmma_bf16(bf16x16 a, bf16x16 b, f32x8 c) {
  // D = A(16x32) * B(32x16) + C(16x16), f32 accumulate
  return __builtin_amdgcn_wmma_f32_16x16x32_bf16(false, a, false, b, (short)0, c,
                                                 false, false);
}

// Load one 16x32 bf16 fragment from a row-major matrix (leading dim `ld` elems).
// Lane n<16: row n, K 0..7 & 16..23 ; lane n>=16: row n-16, K 8..15 & 24..31.
__device__ __forceinline__ bf16x16 frag_ld(const bf16* p, int ld) {
  const int lane = threadIdx.x & 31;
  const bf16* r = p + (size_t)(lane & 15) * ld + ((lane >> 4) << 3);
  bf16x8 lo = *(const bf16x8*)(r);
  bf16x8 hi = *(const bf16x8*)(r + 16);
  return __builtin_shufflevector(lo, hi, 0,1,2,3,4,5,6,7,8,9,10,11,12,13,14,15);
}

// Guarded A-fragment load (zero rows past M).
__device__ __forceinline__ bf16x16 frag_ld_a(const bf16* A, long row0, long M,
                                             int K, int k0) {
  const int lane = threadIdx.x & 31;
  long row = row0 + (lane & 15);
  bf16x16 z = {};
  if (row >= M) return z;
  const bf16* r = A + row * (size_t)K + k0 + ((lane >> 4) << 3);
  bf16x8 lo = *(const bf16x8*)(r);
  bf16x8 hi = *(const bf16x8*)(r + 16);
  return __builtin_shufflevector(lo, hi, 0,1,2,3,4,5,6,7,8,9,10,11,12,13,14,15);
}

#if HAVE_TDM
// Issue one TDM tile load: 32(K, contiguous) x 64(N rows) bf16 tile of BT[N][K]
// into LDS at byte offset `lds_off`. D# built per CDNA5 ISA 8.3/8.4.
__device__ __forceinline__ void tdm_load_b(unsigned lds_off, const bf16* gsrc,
                                           int K, int Nrows) {
  unsigned long long ga = (unsigned long long)(uintptr_t)gsrc;
  u32x4 g0;
  g0.x = 1u;                                   // count=1, user descriptor
  g0.y = lds_off;                              // lds_addr (bytes)
  g0.z = (unsigned)ga;                         // global_addr[31:0]
  g0.w = (unsigned)((ga >> 32) & 0x01FFFFFFu)  // global_addr[56:32]
         | 0x80000000u;                        // type = 2 ("image")
  i32x8 g1;
  g1[0] = 0x00010000;                          // wg_mask=0, data_size=1 (2B)
  g1[1] = (int)((unsigned)(K & 0xFFFF) << 16); // tensor_dim0[15:0]
  g1[2] = (int)(((unsigned)K >> 16) | ((unsigned)(Nrows & 0xFFFF) << 16)); // dim0 hi | dim1 lo
  g1[3] = (int)(((unsigned)Nrows >> 16) | (32u << 16));  // dim1 hi | tile_dim0=32
  g1[4] = 64;                                  // tile_dim1=64, tile_dim2=0
  g1[5] = K;                                   // tensor_dim0_stride[31:0] (elems)
  g1[6] = 0;                                   // stride hi / dim1 stride lo
  g1[7] = 0;
  i32x4 z4 = {0, 0, 0, 0};
#if __clang_major__ >= 23
  i32x8 z8 = {0, 0, 0, 0, 0, 0, 0, 0};
  __builtin_amdgcn_tensor_load_to_lds(g0, g1, z4, z4, z8, 0);
#else
  __builtin_amdgcn_tensor_load_to_lds(g0, g1, z4, z4, 0);
#endif
}
#endif

// ---------------- generic WMMA GEMM ----------------
// C[M,N] = A[M,K](bf16,row-major) * BT[N,K](bf16, i.e. W transposed)
// block = 256 threads = 8 waves; C tile per block = 256 x 64 (32 rows/wave).
// B tile (64x32) double-buffered in LDS via TDM async tensor loads.
template <class Epi>
__global__ __launch_bounds__(256) void gemm_bf16(const bf16* __restrict__ A,
                                                 const bf16* __restrict__ BT,
                                                 int M, int N, int K, Epi epi) {
  __shared__ bf16 Bs[2][64 * 32];
  const int lane = threadIdx.x & 31;
  const int wave = threadIdx.x >> 5;
  const long row0 = (long)blockIdx.x * 256 + wave * 32;
  const long col0 = (long)blockIdx.y * 64;
  const int nk = K >> 5;

  f32x8 acc[2][4] = {};

#if HAVE_TDM
  if (wave == 0)  // prime buffer 0 (TDM ignores EXEC; one wave drives the DMA)
    tdm_load_b((unsigned)(uintptr_t)&Bs[0][0], BT + (size_t)col0 * K, K, N);
#endif

  for (int i = 0; i < nk; ++i) {
    const int cur = i & 1;
    const int k0 = i << 5;
#if HAVE_TDM
    if (wave == 0) {
      if (i + 1 < nk) {  // prefetch next tile into the other buffer
        tdm_load_b((unsigned)(uintptr_t)&Bs[cur ^ 1][0],
                   BT + (size_t)col0 * K + (i + 1) * 32, K, N);
        __builtin_amdgcn_s_wait_tensorcnt(1);  // current tile complete
      } else {
        __builtin_amdgcn_s_wait_tensorcnt(0);
      }
    }
    __syncthreads();
#else
    __syncthreads();
    {  // manual staging fallback: 64(N) x 32(K), 16B per thread
      int r = threadIdx.x >> 2;
      int c = (threadIdx.x & 3) << 3;
      *(bf16x8*)(&Bs[cur][r * 32 + c]) =
          *(const bf16x8*)(&BT[(col0 + r) * (size_t)K + k0 + c]);
      if (k0 + 32 < K)
        __builtin_prefetch(&BT[(col0 + r) * (size_t)K + k0 + 32], 0, 1);
    }
    __syncthreads();
#endif

    bf16x16 a0 = frag_ld_a(A, row0, M, K, k0);
    bf16x16 a1 = frag_ld_a(A, row0 + 16, M, K, k0);
#pragma unroll
    for (int nt = 0; nt < 4; nt++) {
      bf16x16 b = frag_ld(&Bs[cur][(nt * 16) * 32], 32);
      acc[0][nt] = wmma_bf16(a0, b, acc[0][nt]);
      acc[1][nt] = wmma_bf16(a1, b, acc[1][nt]);
    }
    __syncthreads();  // tile consumed; safe to overwrite next iteration
  }

#pragma unroll
  for (int mi = 0; mi < 2; mi++) {
#pragma unroll
    for (int nt = 0; nt < 4; nt++) {
#pragma unroll
      for (int r = 0; r < 8; r++) {
        long row = row0 + mi * 16 + r + ((lane >> 4) << 3);
        if (row < M) epi(row, (int)(col0 + nt * 16 + (lane & 15)), acc[mi][nt][r]);
      }
    }
  }
}

// ---------------- epilogues ----------------
struct EpiQKV {  // scatter qkv -> Q rows, K rows, V^T (for the PV B-operand)
  const float* bias; bf16* Q; bf16* Kb; bf16* Vt;
  __device__ void operator()(long row, int col, float v) const {
    v += bias[col];
    int three = col / DIMC;
    int hc = col - three * DIMC;
    int head = hc >> 6, d = hc & 63;
    int win = (int)(row / NTOK), pos = (int)(row % NTOK);
    long bh = (long)win * HEADS + head;
    if (three == 0)      Q [(bh * NPAD + pos) * HD + d]  = (bf16)v;
    else if (three == 1) Kb[(bh * NPAD + pos) * HD + d]  = (bf16)v;
    else                 Vt[(bh * HD + d) * KPAD + pos]  = (bf16)v;
  }
};

struct EpiProj {  // proj + bias + residual(shortcut from NCHW input) -> x1 (f32)
  const float* bias; const float* hidden; float* x1;
  __device__ void operator()(long row, int col, float v) const {
    int win = (int)(row / NTOK), pos = (int)(row % NTOK);
    int b = win / 25, wr = win % 25;
    int hp = (wr / 5) * WSZ + pos / WSZ;
    int wp = (wr % 5) * WSZ + pos % WSZ;
    if (hp < 64 && wp < 64) {
      long t = ((long)b * 64 + hp) * 64 + wp;
      float sc = hidden[((long)b * DIMC + col) * 4096 + hp * 64 + wp];
      x1[t * DIMC + col] = v + bias[col] + sc;
    }
  }
};

struct EpiGelu {  // fc1 + bias + exact GELU -> bf16
  const float* bias; bf16* Hb;
  __device__ void operator()(long row, int col, float v) const {
    float x = v + bias[col];
    float g = 0.5f * x * (1.0f + erff(x * 0.70710678118654752f));
    Hb[row * (long)MLPD + col] = (bf16)g;
  }
};

struct EpiOut {  // fc2 + bias + residual, store NCHW f32
  const float* bias; const float* x1; float* out;
  __device__ void operator()(long row, int col, float v) const {
    int b = (int)(row >> 12);
    int hw = (int)(row & 4095);
    out[((long)b * DIMC + col) * 4096 + hw] =
        x1[row * (long)DIMC + col] + v + bias[col];
  }
};

// ---------------- elementwise / prep kernels ----------------
__global__ void convertT_kernel(const float* __restrict__ src, bf16* __restrict__ dst,
                                int K, int N) {  // src[K][N] f32 -> dst[N][K] bf16
  long idx = (long)blockIdx.x * 256 + threadIdx.x;
  long total = (long)K * N;
  if (idx >= total) return;
  long n = idx / K, k = idx % K;
  dst[idx] = (bf16)src[k * (long)N + n];
}

__global__ void zero_bf16_kernel(bf16* p, long n) {
  long i = (long)blockIdx.x * blockDim.x + threadIdx.x;
  long st = (long)gridDim.x * blockDim.x;
  for (; i < n; i += st) p[i] = (bf16)0.0f;
}

// LN1 over NCHW input, window-partition (pad to 70x70), bf16 out [19600][768]
__global__ __launch_bounds__(256) void ln1_win_kernel(const float* __restrict__ hs,
                                                      const float* __restrict__ w,
                                                      const float* __restrict__ bv,
                                                      bf16* __restrict__ Xln) {
  long t = blockIdx.x;
  int win = (int)(t / NTOK), pos = (int)(t % NTOK);
  int b = win / 25, wr = win % 25;
  int hp = (wr / 5) * WSZ + pos / WSZ;
  int wp = (wr % 5) * WSZ + pos % WSZ;
  bf16* dst = Xln + t * DIMC;
  if (hp >= 64 || wp >= 64) {  // padded window token: zeros (pad is after LN)
    for (int c = threadIdx.x; c < DIMC; c += 256) dst[c] = (bf16)0.0f;
    return;
  }
  __shared__ float red[256];
  __shared__ float mu_s, rs_s;
  const float* src = hs + (long)b * DIMC * 4096 + hp * 64 + wp;  // stride 4096/chan
  float vals[3];
  float s = 0.f, ss = 0.f;
#pragma unroll
  for (int i = 0; i < 3; i++) {
    int c = threadIdx.x + i * 256;
    float x = src[(long)c * 4096];
    vals[i] = x; s += x; ss += x * x;
  }
  red[threadIdx.x] = s; __syncthreads();
  for (int o = 128; o > 0; o >>= 1) {
    if ((int)threadIdx.x < o) red[threadIdx.x] += red[threadIdx.x + o];
    __syncthreads();
  }
  if (threadIdx.x == 0) mu_s = red[0] * (1.0f / DIMC);
  __syncthreads();
  float mu = mu_s;
  red[threadIdx.x] = ss; __syncthreads();
  for (int o = 128; o > 0; o >>= 1) {
    if ((int)threadIdx.x < o) red[threadIdx.x] += red[threadIdx.x + o];
    __syncthreads();
  }
  if (threadIdx.x == 0) rs_s = rsqrtf(red[0] * (1.0f / DIMC) - mu * mu + EPSLN);
  __syncthreads();
  float rs = rs_s;
#pragma unroll
  for (int i = 0; i < 3; i++) {
    int c = threadIdx.x + i * 256;
    dst[c] = (bf16)((vals[i] - mu) * rs * w[c] + bv[c]);
  }
}

// LN2 over contiguous x1 [16384][768] f32 -> Y bf16
__global__ __launch_bounds__(256) void ln2_kernel(const float* __restrict__ x1,
                                                  const float* __restrict__ w,
                                                  const float* __restrict__ bv,
                                                  bf16* __restrict__ Y) {
  long t = blockIdx.x;
  const float* src = x1 + t * DIMC;
  bf16* dst = Y + t * DIMC;
  __shared__ float red[256];
  __shared__ float mu_s, rs_s;
  float vals[3];
  float s = 0.f, ss = 0.f;
#pragma unroll
  for (int i = 0; i < 3; i++) {
    int c = threadIdx.x + i * 256;
    float x = src[c];
    vals[i] = x; s += x; ss += x * x;
  }
  red[threadIdx.x] = s; __syncthreads();
  for (int o = 128; o > 0; o >>= 1) {
    if ((int)threadIdx.x < o) red[threadIdx.x] += red[threadIdx.x + o];
    __syncthreads();
  }
  if (threadIdx.x == 0) mu_s = red[0] * (1.0f / DIMC);
  __syncthreads();
  float mu = mu_s;
  red[threadIdx.x] = ss; __syncthreads();
  for (int o = 128; o > 0; o >>= 1) {
    if ((int)threadIdx.x < o) red[threadIdx.x] += red[threadIdx.x + o];
    __syncthreads();
  }
  if (threadIdx.x == 0) rs_s = rsqrtf(red[0] * (1.0f / DIMC) - mu * mu + EPSLN);
  __syncthreads();
  float rs = rs_s;
#pragma unroll
  for (int i = 0; i < 3; i++) {
    int c = threadIdx.x + i * 256;
    dst[c] = (bf16)((vals[i] - mu) * rs * w[c] + bv[c]);
  }
}

// decomposed rel-pos: relh[bh][pos][ki] = q(bh,pos) . Rh[i,ki], same for relw
__global__ void relpos_kernel(const bf16* __restrict__ Q,
                              const float* __restrict__ rph,
                              const float* __restrict__ rpw,
                              float* __restrict__ relh, float* __restrict__ relw) {
  long idx = (long)blockIdx.x * 256 + threadIdx.x;
  long total = (long)NBH * NTOK * WSZ;
  if (idx >= total) return;
  int ki = (int)(idx % WSZ);
  long t = idx / WSZ;
  int pos = (int)(t % NTOK);
  long bh = t / NTOK;
  int i = pos / WSZ, j = pos % WSZ;
  const bf16* q = Q + (bh * NPAD + pos) * HD;
  const float* rh = rph + (i - ki + WSZ - 1) * HD;
  const float* rw = rpw + (j - ki + WSZ - 1) * HD;  // ki doubles as kj
  float ah = 0.f, aw = 0.f;
#pragma unroll 4
  for (int d = 0; d < HD; d++) {
    float qv = (float)q[d];
    ah += qv * rh[d];
    aw += qv * rw[d];
  }
  relh[idx] = ah;
  relw[idx] = aw;
}

// ---------------- fused window attention ----------------
// one block per (window, head); scores chunked by 64 query rows in LDS
__global__ __launch_bounds__(256) void attn_kernel(
    const bf16* __restrict__ Q, const bf16* __restrict__ Kb,
    const bf16* __restrict__ Vt, const float* __restrict__ relh,
    const float* __restrict__ relw, bf16* __restrict__ Attn) {
  const int bh = blockIdx.x;
  const int head = bh % HEADS;
  const int win = bh / HEADS;
  const int lane = threadIdx.x & 31;
  const int wave = threadIdx.x >> 5;
  __shared__ bf16 S[64 * 232];  // 64 query rows x 224 key cols (stride 232)

  const bf16* Qb = Q + (long)bh * NPAD * HD;
  const bf16* Kp = Kb + (long)bh * NPAD * HD;
  const bf16* Vp = Vt + (long)bh * HD * KPAD;

  for (int c0 = 0; c0 < NPAD; c0 += 64) {
    const int rows = (NPAD - c0) < 64 ? (NPAD - c0) : 64;
    const int mtiles = rows >> 4;

    // ---- S = scale * Q K^T + bias ----
    for (int job = wave; job < mtiles * 14; job += 8) {
      int mt = job / 14, nt = job % 14;
      f32x8 acc = {};
      if (nt < 13) {
#pragma unroll
        for (int kk = 0; kk < 2; kk++) {
          bf16x16 a = frag_ld(Qb + (size_t)(c0 + mt * 16) * HD + kk * 32, HD);
          bf16x16 b = frag_ld(Kp + (size_t)(nt * 16) * HD + kk * 32, HD);
          acc = wmma_bf16(a, b, acc);
        }
      }
      int colb = nt * 16 + (lane & 15);
#pragma unroll
      for (int r = 0; r < 8; r++) {
        int row = c0 + mt * 16 + r + ((lane >> 4) << 3);
        float v;
        if (row < NTOK && colb < NTOK) {
          long bb = ((long)bh * NTOK + row) * WSZ;
          v = acc[r] * SCALE + relh[bb + colb / WSZ] + relw[bb + colb % WSZ];
        } else {
          v = -1e30f;  // internal padding only; real windows attend everywhere
        }
        S[(row - c0) * 232 + colb] = (bf16)v;
      }
    }
    __syncthreads();

    // ---- softmax rows ----
    if ((int)threadIdx.x < rows) {
      int row = c0 + threadIdx.x;
      bf16* sr = S + threadIdx.x * 232;
      if (row < NTOK) {
        float m = -1e30f;
        for (int k = 0; k < KPAD; k++) m = fmaxf(m, (float)sr[k]);
        float sum = 0.f;
        for (int k = 0; k < KPAD; k++) sum += expf((float)sr[k] - m);
        float inv = 1.0f / sum;
        for (int k = 0; k < KPAD; k++)
          sr[k] = (bf16)(expf((float)sr[k] - m) * inv);
      } else {
        for (int k = 0; k < KPAD; k++) sr[k] = (bf16)0.0f;
      }
    }
    __syncthreads();

    // ---- O = P @ V (P from LDS, V^T from global) ----
    for (int job = wave; job < mtiles * 4; job += 8) {
      int mt = job >> 2, nt = job & 3;
      f32x8 acc = {};
      for (int kk = 0; kk < KPAD; kk += 32) {
        bf16x16 a = frag_ld(S + (size_t)(mt * 16) * 232 + kk, 232);
        bf16x16 b = frag_ld(Vp + (size_t)(nt * 16) * KPAD + kk, KPAD);
        acc = wmma_bf16(a, b, acc);
      }
      int d = nt * 16 + (lane & 15);
#pragma unroll
      for (int r = 0; r < 8; r++) {
        int row = c0 + mt * 16 + r + ((lane >> 4) << 3);
        if (row < NTOK) {
          long tok = (long)win * NTOK + row;
          Attn[tok * DIMC + head * HD + d] = (bf16)acc[r];
        }
      }
    }
    __syncthreads();
  }
}

// ---------------- host launch ----------------
extern "C" void kernel_launch(void* const* d_in, const int* in_sizes, int n_in,
                              void* d_out, int out_size, void* d_ws, size_t ws_size,
                              hipStream_t stream) {
  const float* hs    = (const float*)d_in[0];
  const float* ln1w  = (const float*)d_in[1];
  const float* ln1b  = (const float*)d_in[2];
  const float* qkvw  = (const float*)d_in[3];
  const float* qkvb  = (const float*)d_in[4];
  const float* projw = (const float*)d_in[5];
  const float* projb = (const float*)d_in[6];
  const float* rph   = (const float*)d_in[7];
  const float* rpw   = (const float*)d_in[8];
  const float* ln2w  = (const float*)d_in[9];
  const float* ln2b  = (const float*)d_in[10];
  const float* fc1w  = (const float*)d_in[11];
  const float* fc1b  = (const float*)d_in[12];
  const float* fc2w  = (const float*)d_in[13];
  const float* fc2b  = (const float*)d_in[14];
  float* out = (float*)d_out;

  char* base = (char*)d_ws;
  size_t off = 0;
  auto carve = [&](size_t bytes) -> void* {
    void* p = base + off;
    off += (bytes + 255) & ~(size_t)255;
    return p;
  };
  bf16* Xln   = (bf16*)carve((size_t)MQKV * DIMC * 2);
  bf16* qkvT  = (bf16*)carve((size_t)3 * DIMC * DIMC * 2);
  bf16* projT = (bf16*)carve((size_t)DIMC * DIMC * 2);
  bf16* fc1T  = (bf16*)carve((size_t)MLPD * DIMC * 2);
  bf16* fc2T  = (bf16*)carve((size_t)DIMC * MLPD * 2);
  bf16* Qbuf  = (bf16*)carve((size_t)NBH * NPAD * HD * 2);
  bf16* Kbuf  = (bf16*)carve((size_t)NBH * NPAD * HD * 2);
  bf16* Vtb   = (bf16*)carve((size_t)NBH * HD * KPAD * 2);
  float* relh = (float*)carve((size_t)NBH * NTOK * WSZ * 4);
  float* relw = (float*)carve((size_t)NBH * NTOK * WSZ * 4);
  bf16* Attn  = (bf16*)carve((size_t)MQKV * DIMC * 2);
  float* x1   = (float*)carve((size_t)MMLP * DIMC * 4);
  bf16* Ybuf  = (bf16*)carve((size_t)MMLP * DIMC * 2);
  bf16* Hbuf  = (bf16*)carve((size_t)MMLP * MLPD * 2);

  auto blocks = [](long n) { return (unsigned)((n + 255) / 256); };

  // 1) weights -> transposed bf16
  convertT_kernel<<<blocks((long)DIMC * 3 * DIMC), 256, 0, stream>>>(qkvw, qkvT, DIMC, 3 * DIMC);
  convertT_kernel<<<blocks((long)DIMC * DIMC), 256, 0, stream>>>(projw, projT, DIMC, DIMC);
  convertT_kernel<<<blocks((long)DIMC * MLPD), 256, 0, stream>>>(fc1w, fc1T, DIMC, MLPD);
  convertT_kernel<<<blocks((long)MLPD * DIMC), 256, 0, stream>>>(fc2w, fc2T, MLPD, DIMC);

  // 2) zero-pad Q/K/V^T buffers (pad rows/cols must be zero)
  zero_bf16_kernel<<<2048, 256, 0, stream>>>(Qbuf, (long)NBH * NPAD * HD);
  zero_bf16_kernel<<<2048, 256, 0, stream>>>(Kbuf, (long)NBH * NPAD * HD);
  zero_bf16_kernel<<<2048, 256, 0, stream>>>(Vtb, (long)NBH * HD * KPAD);

  // 3) LN1 + window partition
  ln1_win_kernel<<<MQKV, 256, 0, stream>>>(hs, ln1w, ln1b, Xln);

  // 4) QKV GEMM (19600 x 2304 x 768)
  {
    EpiQKV e{qkvb, Qbuf, Kbuf, Vtb};
    gemm_bf16<EpiQKV><<<dim3((MQKV + 255) / 256, (3 * DIMC) / 64), 256, 0, stream>>>(
        Xln, qkvT, MQKV, 3 * DIMC, DIMC, e);
  }

  // 5) decomposed rel-pos bias vectors
  relpos_kernel<<<blocks((long)NBH * NTOK * WSZ), 256, 0, stream>>>(Qbuf, rph, rpw,
                                                                    relh, relw);

  // 6) fused attention per (window, head)
  attn_kernel<<<NBH, 256, 0, stream>>>(Qbuf, Kbuf, Vtb, relh, relw, Attn);

  // 7) proj GEMM + residual -> x1
  {
    EpiProj e{projb, hs, x1};
    gemm_bf16<EpiProj><<<dim3((MQKV + 255) / 256, DIMC / 64), 256, 0, stream>>>(
        Attn, projT, MQKV, DIMC, DIMC, e);
  }

  // 8) LN2
  ln2_kernel<<<MMLP, 256, 0, stream>>>(x1, ln2w, ln2b, Ybuf);

  // 9) fc1 + GELU
  {
    EpiGelu e{fc1b, Hbuf};
    gemm_bf16<EpiGelu><<<dim3(MMLP / 256, MLPD / 64), 256, 0, stream>>>(
        Ybuf, fc1T, MMLP, MLPD, DIMC, e);
  }

  // 10) fc2 + residual -> NCHW output
  {
    EpiOut e{fc2b, x1, out};
    gemm_bf16<EpiOut><<<dim3(MMLP / 256, DIMC / 64), 256, 0, stream>>>(
        Hbuf, fc2T, MMLP, DIMC, MLPD, e);
  }
}
